// UniTRep_71116068487915
// MI455X (gfx1250) — matrix operand order repaired and checked by reading, not compiled, not run on hardware
//
#include <hip/hip_runtime.h>
#include <hip/hip_bf16.h>
#include <math.h>

#define B_ 256
#define L_ 20
#define H_ 128
#define HEADS_ 8
#define NT_ 4
#define NR_ 4
#define CHILD_ 30001
#define PARENT_ 500
#define N1S_ 50000
#define N1D_ 20000
#define E1_ 300000
#define N2_ 20000
#define E2_ 200000
#define T_ (B_ * L_)
#define N4D_ 5000
#define E4_ 150000
#define EC2P_ 8000

typedef __attribute__((ext_vector_type(16))) __bf16 v16bf;
typedef __attribute__((ext_vector_type(8))) float v8f;

struct U16B { uint4 lo, hi; };  // 32 bytes = 16 bf16

// ---------------- device helpers ----------------

__device__ __forceinline__ unsigned short f2bf_rte(float f) {
  unsigned u = __float_as_uint(f);
  unsigned r = u + 0x7FFFu + ((u >> 16) & 1u);
  return (unsigned short)(r >> 16);
}

__device__ __forceinline__ void atomicMaxF(float* addr, float val) {
  if (val >= 0.0f)
    atomicMax((int*)addr, __float_as_int(val));
  else
    atomicMin((unsigned int*)addr, __float_as_uint(val));
}

__device__ __forceinline__ float sigf(float x) { return 1.0f / (1.0f + expf(-x)); }

// A fragment: 16x32 bf16 from row-major bf16 buffer, layout per ISA 7.12.2.
// Row clamped (not masked) -> no exec divergence in the K loop.
__device__ __forceinline__ v16bf load_a_frag(const unsigned short* A, int lda, int M,
                                             int row0, int kk, int lane) {
  int m = row0 + (lane & 15);
  if (m >= M) m = M - 1;
  int kb = kk + ((lane >> 4) << 3);
  const uint4* p = (const uint4*)(A + m * lda + kb);
  U16B t;
  t.lo = p[0];   // K = kb .. kb+7
  t.hi = p[2];   // K = kb+16 .. kb+23
  return __builtin_bit_cast(v16bf, t);
}

// B fragment from pre-packed weights: one 32B contiguous chunk per lane.
// Packed index: ((tile*KT + ks)*32 + lane)*16 + e
__device__ __forceinline__ v16bf load_b_packed(const unsigned short* Wp, int tile,
                                               int KT, int ks, int lane) {
  const uint4* p = (const uint4*)(Wp + (((tile * KT + ks) << 5) + lane) * 16);
  U16B t;
  t.lo = p[0];
  t.hi = p[1];
  return __builtin_bit_cast(v16bf, t);
}

// ---------------- weight packer ----------------
// Packs nmat row-major f32 (K,N) matrices into WMMA-B-fragment bf16 layout,
// zero-padding K up to Kpad (multiple of 32).
__global__ void k_pack_wb(unsigned short* out, const float* W, int K, int Kpad, int N,
                          int nmat) {
  int i = blockIdx.x * 256 + threadIdx.x;
  int KT = Kpad >> 5;
  int per = (N >> 4) * KT * 512;
  if (i >= nmat * per) return;
  int m = i / per;
  int r = i - m * per;
  int e = r & 15;
  int lane = (r >> 4) & 31;
  int ks = (r >> 9) % KT;
  int tile = r / (KT * 512);
  int k = (ks << 5) + ((lane >> 4) << 3) + (e & 7) + ((e >> 3) << 4);
  int n = (tile << 4) + (lane & 15);
  float v = (k < K) ? W[(size_t)m * K * N + (size_t)k * N + n] : 0.0f;
  out[i] = f2bf_rte(v);
}

// ---------------- GEMM kernels (WMMA bf16 -> f32) ----------------

// out[M,N] = act(A_bf16[M,Kpad] @ Wp + bias); Wp pre-packed, Kpad%32==0.
// 8 waves/block, wave w handles n-tile (blockIdx.y*8 + w).
__global__ __launch_bounds__(256) void k_gemm(
    const unsigned short* __restrict__ A, const unsigned short* __restrict__ Wp,
    const float* __restrict__ bias, float* __restrict__ outF,
    unsigned short* __restrict__ outBF, int M, int N, int Kpad, int lda, int ldo,
    int act) {
  int lane = threadIdx.x & 31;
  int wave = threadIdx.x >> 5;
  int row0 = blockIdx.x << 4;
  int tile = blockIdx.y * 8 + wave;
  int c0 = tile << 4;
  if (c0 >= N) return;  // wave-uniform
  int n = c0 + (lane & 15);
  int KT = Kpad >> 5;
  v8f acc;
#pragma unroll
  for (int j = 0; j < 8; ++j) acc[j] = 0.0f;
  for (int ks = 0; ks < KT; ++ks) {
    v16bf a = load_a_frag(A, lda, M, row0, ks << 5, lane);
    v16bf b = load_b_packed(Wp, tile, KT, ks, lane);
    acc = __builtin_amdgcn_wmma_f32_16x16x32_bf16(false, a, false, b, (short)0, acc,
                                                  false, false);
  }
  float bi = bias ? bias[n] : 0.0f;
  int mh = (lane >> 4) << 3;
#pragma unroll
  for (int j = 0; j < 8; ++j) {
    int m = row0 + mh + j;
    if (m < M) {
      float v = acc[j] + bi;
      if (act == 1) v = fmaxf(v, 0.0f);
      else if (act == 2) v = tanhf(v);
      if (outF) outF[(size_t)m * ldo + n] = v;
      if (outBF) outBF[(size_t)m * ldo + n] = f2bf_rte(v);
    }
  }
}

// Typed GEMM (K=N=128): out[m,:] = act(A[m,:] @ W[type[m]] + residual[m,:])
// Wp4 = 4 packed type matrices, stride 16384 elems; 4 WMMA accumulators.
__global__ __launch_bounds__(256) void k_gemm_typed(
    const unsigned short* __restrict__ A, const unsigned short* __restrict__ Wp4,
    const int* __restrict__ rtype, const float* __restrict__ residual,
    float* __restrict__ outF, unsigned short* __restrict__ outBF, int M, int act) {
  const int KT = 4, PER = 16384;
  int lane = threadIdx.x & 31;
  int wave = threadIdx.x >> 5;
  int row0 = blockIdx.x << 4;
  int tile = wave;
  int n = (tile << 4) + (lane & 15);
  v8f acc0, acc1, acc2, acc3;
#pragma unroll
  for (int j = 0; j < 8; ++j) { acc0[j] = 0.f; acc1[j] = 0.f; acc2[j] = 0.f; acc3[j] = 0.f; }
#pragma unroll
  for (int ks = 0; ks < KT; ++ks) {
    v16bf a = load_a_frag(A, 128, M, row0, ks << 5, lane);
    v16bf b0 = load_b_packed(Wp4 + 0 * PER, tile, KT, ks, lane);
    acc0 = __builtin_amdgcn_wmma_f32_16x16x32_bf16(false, a, false, b0, (short)0, acc0, false, false);
    v16bf b1 = load_b_packed(Wp4 + 1 * PER, tile, KT, ks, lane);
    acc1 = __builtin_amdgcn_wmma_f32_16x16x32_bf16(false, a, false, b1, (short)0, acc1, false, false);
    v16bf b2 = load_b_packed(Wp4 + 2 * PER, tile, KT, ks, lane);
    acc2 = __builtin_amdgcn_wmma_f32_16x16x32_bf16(false, a, false, b2, (short)0, acc2, false, false);
    v16bf b3 = load_b_packed(Wp4 + 3 * PER, tile, KT, ks, lane);
    acc3 = __builtin_amdgcn_wmma_f32_16x16x32_bf16(false, a, false, b3, (short)0, acc3, false, false);
  }
  int mh = (lane >> 4) << 3;
#pragma unroll
  for (int j = 0; j < 8; ++j) {
    int m = row0 + mh + j;
    if (m < M) {
      int t = rtype[m];
      float v = (t == 0) ? acc0[j] : (t == 1) ? acc1[j] : (t == 2) ? acc2[j] : acc3[j];
      if (residual) v += residual[(size_t)m * 128 + n];
      if (act == 1) v = fmaxf(v, 0.0f);
      else if (act == 2) v = tanhf(v);
      if (outF) outF[(size_t)m * 128 + n] = v;
      if (outBF) outBF[(size_t)m * 128 + n] = f2bf_rte(v);
    }
  }
}

// ---------------- elementwise / gather kernels ----------------

__global__ void k_fill(float* p, float v, int n) {
  int i = blockIdx.x * 256 + threadIdx.x;
  if (i < n) p[i] = v;
}

__global__ void k_f2bf(unsigned short* out, const float* in, int n) {
  int i = blockIdx.x * 256 + threadIdx.x;
  if (i < n) out[i] = f2bf_rte(in[i]);
}

__global__ void k_gather_rows(float* out, const float* src, const int* idx, int n) {
  int i = blockIdx.x * 256 + threadIdx.x;
  if (i >= n * 128) return;
  int r = i >> 7, d = i & 127;
  out[i] = src[(size_t)idx[r] * 128 + d];
}

__global__ void k_gather_int(int* out, const int* src, const int* idx, int n) {
  int i = blockIdx.x * 256 + threadIdx.x;
  if (i < n) out[i] = src[idx[i]];
}

__global__ void k_tok_transpose_int(int* out, const int* in) {
  int i = blockIdx.x * 256 + threadIdx.x;
  if (i >= T_) return;
  int b = i / L_, l = i % L_;
  out[l * B_ + b] = in[i];
}

// Build (B*L, 288) bf16 row = [st(128) | et(128) | tv(16) | zero-pad(16)]
__global__ void k_build_xcat(unsigned short* xcat, const int* seq, const int* dur,
                             const int* stime, const int* etime, const float* emb,
                             const float* dia_w, const float* dia_b,
                             const float* t2v_w, const float* t2v_b) {
  int i = blockIdx.x * 256 + threadIdx.x;
  if (i >= T_ * 288) return;
  int r = i / 288, j = i % 288;
  float val = 0.0f;
  int sid = seq[r];
  if (j < 256) {
    int d = j & 127;
    float e0 = emb[(size_t)sid * 128 + d];
    if (d < 64) {
      val = e0;
    } else {
      float t = (float)((j < 128) ? stime[r] : etime[r]);
      val = e0 * sinf(dia_w[(size_t)sid * 128 + d] * t + dia_b[(size_t)sid * 128 + d]);
    }
  } else if (j < 272) {
    int jj = j - 256;
    float t = (float)dur[r];
    val = (jj == 0) ? (t2v_w[0] * t + t2v_b[0]) : sinf(t2v_w[jj] * t + t2v_b[jj]);
  }
  xcat[(size_t)r * 288 + j] = f2bf_rte(val);
}

// LayerNorm (+ optional time-modulation) -> bf16.  Warp per row.
__global__ __launch_bounds__(256) void k_tmod_ln(unsigned short* out, const float* src,
                                                 const float* dw, const float* db,
                                                 const int* dia_idx, int n, int mode) {
  int r = blockIdx.x * 8 + (threadIdx.x >> 5);
  int lane = threadIdx.x & 31;
  if (r >= n) return;
  const float* x = src + (size_t)r * 128;
  float v[4], s = 0.f, ss = 0.f;
#pragma unroll
  for (int i = 0; i < 4; ++i) {
    float t = x[lane + 32 * i];
    v[i] = t; s += t; ss += t * t;
  }
  for (int off = 16; off; off >>= 1) {
    s += __shfl_xor(s, off, 32);
    ss += __shfl_xor(ss, off, 32);
  }
  float mean = s * (1.0f / 128.0f);
  float var = ss * (1.0f / 128.0f) - mean * mean;
  float rs = rsqrtf(var + 1e-5f);
  int di = (mode == 1) ? dia_idx[r] : 0;
#pragma unroll
  for (int i = 0; i < 4; ++i) {
    int d = lane + 32 * i;
    float y = (v[i] - mean) * rs;
    if (mode == 1 && d >= 64)
      y *= sinf(dw[(size_t)di * 128 + d] + db[(size_t)di * 128 + d]);
    out[(size_t)r * 128 + d] = f2bf_rte(y);
  }
}

// LSTM pointwise step. rowx = b*sb + l*sl into gates_x.
__global__ void k_lstm_step(const float* gx, const float* gh, float* c,
                            unsigned short* hbf, float* outA, float* outB, int l,
                            int sb, int sl) {
  int i = blockIdx.x * 256 + threadIdx.x;
  if (i >= B_ * 128) return;
  int b = i >> 7, d = i & 127;
  size_t rx = (size_t)(b * sb + l * sl) * 512;
  size_t rh = (size_t)b * 512;
  float gi = gx[rx + d] + gh[rh + d];
  float gf = gx[rx + 128 + d] + gh[rh + 128 + d];
  float gg = gx[rx + 256 + d] + gh[rh + 256 + d];
  float go = gx[rx + 384 + d] + gh[rh + 384 + d];
  float cc = sigf(gf) * c[i] + sigf(gi) * tanhf(gg);
  float h = sigf(go) * tanhf(cc);
  c[i] = cc;
  hbf[i] = f2bf_rte(h);
  outA[((size_t)l * B_ + b) * 128 + d] = h;                // time-major (l*B + b)
  if (outB) outB[((size_t)b * L_ + l) * 128 + d] = h;      // batch-major (b*L + l)
}

// ---------------- HGT edge kernels ----------------

__global__ void k_edge_score(float* sc, float* mmax, const float* q, const float* k,
                             const float* P, const int* esrc, const int* edst,
                             const int* etype, const int* ew, const float* mu, int E) {
  int i = blockIdx.x * 256 + threadIdx.x;
  if (i >= E * 8) return;
  int e = i >> 3, h = i & 7;
  int s = esrc[e], d = edst[e];
  const float* qp = q + (size_t)d * 128 + h * 16;
  const float* kp = k + (size_t)s * 128 + h * 16;
  const float* pp = P + (size_t)ew[e] * 128 + h * 16;
  float acc = 0.f;
#pragma unroll
  for (int t = 0; t < 16; ++t) acc += qp[t] * (kp[t] + pp[t]);
  acc *= mu[etype[e] * 8 + h] * 0.25f;  // 1/sqrt(DK)=0.25
  sc[i] = acc;
  atomicMaxF(&mmax[(size_t)d * 8 + h], acc);
}

__global__ void k_edge_exp(float* sc, const float* mmax, float* den, const int* edst,
                           int E) {
  int i = blockIdx.x * 256 + threadIdx.x;
  if (i >= E * 8) return;
  int e = i >> 3, h = i & 7;
  int d = edst[e];
  float m = mmax[(size_t)d * 8 + h];
  if (!isfinite(m)) m = 0.f;
  float ex = expf(sc[i] - m);
  sc[i] = ex;
  atomicAdd(&den[(size_t)d * 8 + h], ex);
}

__global__ void k_edge_agg(float* agg, const float* sc, const float* den,
                           const float* v, const int* esrc, const int* edst, int E) {
  int i = blockIdx.x * 256 + threadIdx.x;
  if (i >= E * 128) return;
  int e = i >> 7, dd = i & 127;
  int h = dd >> 4;
  int d = edst[e];
  float a = sc[(size_t)e * 8 + h] / fmaxf(den[(size_t)d * 8 + h], 1e-9f);
  atomicAdd(&agg[(size_t)d * 128 + dd], a * v[(size_t)esrc[e] * 128 + dd]);
}

// ---------------- misc pipeline kernels ----------------

__global__ void k_dsth2(float* out, const float* xg, const float* his,
                        const int* tok_sel) {
  int i = blockIdx.x * 256 + threadIdx.x;
  if (i >= T_ * 128) return;
  int r = i >> 7, d = i & 127;
  out[i] = xg[(size_t)tok_sel[r] * 128 + d] + his[i];
}

__global__ __launch_bounds__(256) void k_row_dot_ia(float* a, const float* inner,
                                                    const float* iaW,
                                                    const int* tok_nt, int n) {
  int r = blockIdx.x * 8 + (threadIdx.x >> 5);
  int lane = threadIdx.x & 31;
  if (r >= n) return;
  int t = tok_nt[r];
  float s = 0.f;
#pragma unroll
  for (int i = 0; i < 4; ++i) {
    int d = lane + 32 * i;
    s += inner[(size_t)r * 128 + d] * iaW[t * 128 + d];
  }
  for (int off = 16; off; off >>= 1) s += __shfl_xor(s, off, 32);
  if (lane == 0) a[r] = s;
}

__global__ void k_seg_max(float* m, const float* a, const int* seg, int n) {
  int i = blockIdx.x * 256 + threadIdx.x;
  if (i < n) atomicMaxF(&m[seg[i]], a[i]);
}

__global__ void k_seg_exp(float* a, const float* m, float* den, const int* seg, int n) {
  int i = blockIdx.x * 256 + threadIdx.x;
  if (i >= n) return;
  float mm = m[seg[i]];
  if (!isfinite(mm)) mm = 0.f;
  float ex = expf(a[i] - mm);
  a[i] = ex;
  atomicAdd(&den[seg[i]], ex);
}

__global__ void k_token_accum(float* node_embed, float* cnt, const float* ex,
                              const float* den, const float* ent, const int* tok_id) {
  int i = blockIdx.x * 256 + threadIdx.x;
  if (i >= T_ * 128) return;
  int r = i >> 7, d = i & 127;
  int c = tok_id[r];
  float w = ex[r] / fmaxf(den[c], 1e-9f);
  atomicAdd(&node_embed[(size_t)c * 128 + d], w * ent[i]);
  if (d == 0) atomicAdd(&cnt[c], 1.0f);
}

__global__ void k_x0(float* x0, const float* emb, const float* node_embed,
                     const float* cnt, const float* hist_emb, const float* skip,
                     const int* g1_nid, const int* g1_ntype, int n) {
  int i = blockIdx.x * 256 + threadIdx.x;
  if (i >= n * 128) return;
  int r = i >> 7, d = i & 127;
  int c = g1_nid[r];
  float he = (cnt[c] > 0.f) ? node_embed[(size_t)c * 128 + d]
                            : hist_emb[(size_t)c * 128 + d];
  float al = sigf(skip[g1_ntype[r]]);
  x0[i] = emb[(size_t)c * 128 + d] * al + he * (1.0f - al);
}

__global__ void k_parent(float* psum, float* pcnt, const float* child,
                         const int* pe_p, const int* pe_c) {
  int i = blockIdx.x * 256 + threadIdx.x;
  if (i >= EC2P_ * 128) return;
  int e = i >> 7, d = i & 127;
  atomicAdd(&psum[(size_t)pe_p[e] * 128 + d], child[(size_t)pe_c[e] * 128 + d]);
  if (d == 0) atomicAdd(&pcnt[pe_p[e]], 1.0f);
}

__global__ void k_parent_div(float* out, const float* psum, const float* pcnt) {
  int i = blockIdx.x * 256 + threadIdx.x;
  if (i >= PARENT_ * 128) return;
  out[i] = psum[i] / fmaxf(pcnt[i >> 7], 1.0f);
}

// ---------------- host orchestration ----------------

static inline unsigned cdiv(unsigned a, unsigned b) { return (a + b - 1) / b; }

static void run_htgt(hipStream_t s, int Ns, int Nd, int E, const float* src_h,
                     const int* src_t, const int* dia_idx, const int* esrc,
                     const int* edst, const int* etype, const int* ew,
                     const float* dst_h, const int* dst_t,
                     const unsigned short* KWp, const unsigned short* QWp,
                     const unsigned short* VWp, const unsigned short* OWp,
                     const unsigned short* EWp, const float* mu, const float* dia_w,
                     const float* dia_b, unsigned short* bf_src,
                     unsigned short* bf_dst, unsigned short* bf_ee, float* kbuf,
                     float* vbuf, float* qbuf, float* Ptab, float* scores,
                     float* mmax, float* denb, float* agg, unsigned short* bf_agg,
                     float* outF, unsigned short* outBF, int act) {
  k_tmod_ln<<<cdiv(Ns, 8), 256, 0, s>>>(bf_src, src_h, dia_w, dia_b, dia_idx, Ns, 1);
  k_gemm_typed<<<cdiv(Ns, 16), 256, 0, s>>>(bf_src, KWp, src_t, nullptr, kbuf, nullptr, Ns, 0);
  k_gemm_typed<<<cdiv(Ns, 16), 256, 0, s>>>(bf_src, VWp, src_t, nullptr, vbuf, nullptr, Ns, 0);
  k_tmod_ln<<<cdiv(Nd, 8), 256, 0, s>>>(bf_dst, dst_h, nullptr, nullptr, nullptr, Nd, 0);
  k_gemm_typed<<<cdiv(Nd, 16), 256, 0, s>>>(bf_dst, QWp, dst_t, nullptr, qbuf, nullptr, Nd, 0);
  k_gemm<<<dim3(cdiv(50, 16), 1), 256, 0, s>>>(bf_ee, EWp, nullptr, Ptab, nullptr, 50,
                                               128, 32, 32, 128, 0);
  k_fill<<<cdiv(Nd * 8, 256), 256, 0, s>>>(mmax, -__builtin_huge_valf(), Nd * 8);
  k_fill<<<cdiv(Nd * 8, 256), 256, 0, s>>>(denb, 0.f, Nd * 8);
  k_fill<<<cdiv(Nd * 128, 256), 256, 0, s>>>(agg, 0.f, Nd * 128);
  k_edge_score<<<cdiv(E * 8, 256), 256, 0, s>>>(scores, mmax, qbuf, kbuf, Ptab, esrc,
                                                edst, etype, ew, mu, E);
  k_edge_exp<<<cdiv(E * 8, 256), 256, 0, s>>>(scores, mmax, denb, edst, E);
  k_edge_agg<<<cdiv(E * 128, 256), 256, 0, s>>>(agg, scores, denb, vbuf, esrc, edst, E);
  k_f2bf<<<cdiv(Nd * 128, 256), 256, 0, s>>>(bf_agg, agg, Nd * 128);
  k_gemm_typed<<<cdiv(Nd, 16), 256, 0, s>>>(bf_agg, OWp, dst_t, dst_h, outF, outBF, Nd, act);
}

extern "C" void kernel_launch(void* const* d_in, const int* in_sizes, int n_in,
                              void* d_out, int out_size, void* d_ws, size_t ws_size,
                              hipStream_t stream) {
  (void)in_sizes; (void)n_in; (void)out_size; (void)ws_size;
  const int* seq = (const int*)d_in[0];
  const int* seq_ntype = (const int*)d_in[1];
  const int* dur = (const int*)d_in[2];
  const int* stime = (const int*)d_in[3];
  const int* etime = (const int*)d_in[4];
  const int* g1_nid = (const int*)d_in[5];
  const int* g1_ntype = (const int*)d_in[6];
  const int* g1_esrc = (const int*)d_in[7];
  const int* g1_edst = (const int*)d_in[8];
  const int* g1_etype = (const int*)d_in[9];
  const int* g1_ew = (const int*)d_in[10];
  const int* g2_src_sel = (const int*)d_in[11];
  const int* g2_esrc = (const int*)d_in[12];
  const int* g2_edst = (const int*)d_in[13];
  const int* g2_etype = (const int*)d_in[14];
  const int* g2_ew = (const int*)d_in[15];
  const int* tok_sel = (const int*)d_in[16];
  const int* g4_esrc = (const int*)d_in[17];
  const int* g4_edst = (const int*)d_in[18];
  const int* g4_etype = (const int*)d_in[19];
  const int* g4_ew = (const int*)d_in[20];
  const int* pe_p = (const int*)d_in[21];
  const int* pe_c = (const int*)d_in[22];
  const float* emb = (const float*)d_in[23];
  const float* hist_emb = (const float*)d_in[24];
  const float* edge_emb = (const float*)d_in[25];
  const float* dia_w = (const float*)d_in[26];
  const float* dia_b = (const float*)d_in[27];
  const float* t2v_w = (const float*)d_in[28];
  const float* t2v_b = (const float*)d_in[29];
  const float* exp_W = (const float*)d_in[30];
  const float* exp_b = (const float*)d_in[31];
  const float* lstm1_Wi = (const float*)d_in[32];
  const float* lstm1_Wh = (const float*)d_in[33];
  const float* lstm1_b = (const float*)d_in[34];
  const float* lstm2_Wi = (const float*)d_in[35];
  const float* lstm2_Wh = (const float*)d_in[36];
  const float* lstm2_b = (const float*)d_in[37];
  const float* convK = (const float*)d_in[38];
  const float* convQ = (const float*)d_in[39];
  const float* convV = (const float*)d_in[40];
  const float* convO = (const float*)d_in[41];
  const float* convE = (const float*)d_in[42];
  const float* convMu = (const float*)d_in[43];
  const float* ipW = (const float*)d_in[44];
  const float* iaW = (const float*)d_in[45];
  const float* skip = (const float*)d_in[46];

  float* out = (float*)d_out;
  float* out_h2 = out;                                // (B,L,H)
  float* out_child = out + (size_t)T_ * H_;           // (N4D,H)
  float* out_parent = out_child + (size_t)N4D_ * H_;  // (PARENT,H)

  // ---- workspace carve-out ----
  char* wsb = (char*)d_ws;
  size_t off = 0;
  auto alloc = [&](size_t elems, size_t esz) -> void* {
    void* p = wsb + off;
    off += ((elems * esz + 255) / 256) * 256;
    return p;
  };
  unsigned short* bf_xcat = (unsigned short*)alloc((size_t)T_ * 288, 2);
  unsigned short* bf_x = (unsigned short*)alloc((size_t)T_ * H_, 2);
  float* gates_x = (float*)alloc((size_t)T_ * 512, 4);
  float* gates_h = (float*)alloc((size_t)B_ * 512, 4);
  unsigned short* hbf = (unsigned short*)alloc((size_t)B_ * H_, 2);
  float* cbuf = (float*)alloc((size_t)B_ * H_, 4);
  float* his = (float*)alloc((size_t)T_ * H_, 4);
  float* ent = (float*)alloc((size_t)T_ * H_, 4);
  unsigned short* bf_ent = (unsigned short*)alloc((size_t)T_ * H_, 2);
  float* srcbuf = (float*)alloc((size_t)N1S_ * H_, 4);
  unsigned short* bf_src = (unsigned short*)alloc((size_t)N1S_ * H_, 2);
  float* kbuf = (float*)alloc((size_t)N1S_ * H_, 4);
  float* vbuf = (float*)alloc((size_t)N1S_ * H_, 4);
  float* qbuf = (float*)alloc((size_t)N1D_ * H_, 4);
  unsigned short* bf_dst = (unsigned short*)alloc((size_t)N1D_ * H_, 2);
  float* Ptab = (float*)alloc(50 * H_, 4);
  unsigned short* bf_ee = (unsigned short*)alloc(50 * 32, 2);
  float* scores = (float*)alloc((size_t)E1_ * HEADS_, 4);
  float* mmax = (float*)alloc((size_t)N1D_ * HEADS_, 4);
  float* denb = (float*)alloc((size_t)N1D_ * HEADS_, 4);
  float* agg = (float*)alloc((size_t)N1D_ * H_, 4);
  unsigned short* bf_agg = (unsigned short*)alloc((size_t)N1D_ * H_, 2);
  float* xg = (float*)alloc((size_t)N1D_ * H_, 4);
  int* gnid2 = (int*)alloc(N2_, 4);
  int* srct2 = (int*)alloc(N2_, 4);
  float* dsth2 = (float*)alloc((size_t)T_ * H_, 4);
  unsigned short* bf_x2 = (unsigned short*)alloc((size_t)T_ * H_, 2);
  int* tok_id = (int*)alloc(T_, 4);
  int* tok_nt = (int*)alloc(T_, 4);
  float* inner = (float*)alloc((size_t)T_ * H_, 4);
  float* abuf = (float*)alloc(T_, 4);
  float* mchild = (float*)alloc(CHILD_, 4);
  float* dchild = (float*)alloc(CHILD_, 4);
  float* cnt = (float*)alloc(CHILD_, 4);
  float* node_embed = (float*)alloc((size_t)CHILD_ * H_, 4);
  float* x3 = (float*)alloc((size_t)N1D_ * H_, 4);
  float* psum = (float*)alloc((size_t)PARENT_ * H_, 4);
  float* pcnt = (float*)alloc(PARENT_, 4);
  // packed weights (bf16, WMMA B-fragment layout)
  const int PER128 = 16384;          // one 128x128 matrix packed
  const int PER512 = 65536;          // one 128x512 matrix packed
  const int PEREXP = 8 * 9 * 512;    // 272(->288) x 128
  const int PERE = 8 * 1 * 512;      // 32 x 128
  unsigned short* expWp = (unsigned short*)alloc(PEREXP, 2);
  unsigned short* l1WiP = (unsigned short*)alloc(PER512, 2);
  unsigned short* l1WhP = (unsigned short*)alloc(PER512, 2);
  unsigned short* l2WiP = (unsigned short*)alloc(PER512, 2);
  unsigned short* l2WhP = (unsigned short*)alloc(PER512, 2);
  unsigned short* convKP = (unsigned short*)alloc((size_t)16 * PER128, 2);
  unsigned short* convQP = (unsigned short*)alloc((size_t)16 * PER128, 2);
  unsigned short* convVP = (unsigned short*)alloc((size_t)16 * PER128, 2);
  unsigned short* convOP = (unsigned short*)alloc((size_t)16 * PER128, 2);
  unsigned short* convEP = (unsigned short*)alloc((size_t)4 * PERE, 2);
  unsigned short* ipWP = (unsigned short*)alloc((size_t)4 * PER128, 2);

  // ---- phase 0: pack all weights to WMMA-B bf16 fragments ----
  k_pack_wb<<<cdiv(PEREXP, 256), 256, 0, stream>>>(expWp, exp_W, 272, 288, 128, 1);
  k_pack_wb<<<cdiv(PER512, 256), 256, 0, stream>>>(l1WiP, lstm1_Wi, 128, 128, 512, 1);
  k_pack_wb<<<cdiv(PER512, 256), 256, 0, stream>>>(l1WhP, lstm1_Wh, 128, 128, 512, 1);
  k_pack_wb<<<cdiv(PER512, 256), 256, 0, stream>>>(l2WiP, lstm2_Wi, 128, 128, 512, 1);
  k_pack_wb<<<cdiv(PER512, 256), 256, 0, stream>>>(l2WhP, lstm2_Wh, 128, 128, 512, 1);
  k_pack_wb<<<cdiv(16 * PER128, 256), 256, 0, stream>>>(convKP, convK, 128, 128, 128, 16);
  k_pack_wb<<<cdiv(16 * PER128, 256), 256, 0, stream>>>(convQP, convQ, 128, 128, 128, 16);
  k_pack_wb<<<cdiv(16 * PER128, 256), 256, 0, stream>>>(convVP, convV, 128, 128, 128, 16);
  k_pack_wb<<<cdiv(16 * PER128, 256), 256, 0, stream>>>(convOP, convO, 128, 128, 128, 16);
  k_pack_wb<<<cdiv(4 * PERE, 256), 256, 0, stream>>>(convEP, convE, 32, 32, 128, 4);
  k_pack_wb<<<cdiv(4 * PER128, 256), 256, 0, stream>>>(ipWP, ipW, 128, 128, 128, 4);

  // ---- phase A: features -> x (bf16) ----
  k_f2bf<<<cdiv(50 * 32, 256), 256, 0, stream>>>(bf_ee, edge_emb, 50 * 32);
  k_build_xcat<<<cdiv(T_ * 288, 256), 256, 0, stream>>>(bf_xcat, seq, dur, stime, etime,
                                                        emb, dia_w, dia_b, t2v_w, t2v_b);
  k_gemm<<<dim3(cdiv(T_, 16), 1), 256, 0, stream>>>(bf_xcat, expWp, exp_b, nullptr, bf_x,
                                                    T_, 128, 288, 288, 128, 1);

  // ---- phase B: LSTM1 -> his (time-major) ----
  k_gemm<<<dim3(cdiv(T_, 16), 4), 256, 0, stream>>>(bf_x, l1WiP, lstm1_b, gates_x,
                                                    nullptr, T_, 512, 128, 128, 512, 0);
  k_fill<<<cdiv(B_ * H_ / 2, 256), 256, 0, stream>>>((float*)hbf, 0.f, B_ * H_ / 2);
  k_fill<<<cdiv(B_ * H_, 256), 256, 0, stream>>>(cbuf, 0.f, B_ * H_);
  for (int l = 0; l < L_; ++l) {
    k_gemm<<<dim3(cdiv(B_, 16), 4), 256, 0, stream>>>(hbf, l1WhP, nullptr, gates_h,
                                                      nullptr, B_, 512, 128, 128, 512, 0);
    k_lstm_step<<<cdiv(B_ * H_, 256), 256, 0, stream>>>(gates_x, gates_h, cbuf, hbf, his,
                                                        nullptr, l, L_, 1);
  }

  // ---- phase C: HGT layer 0 -> xg ----
  k_gather_rows<<<cdiv(N1S_ * 128, 256), 256, 0, stream>>>(srcbuf, emb, g1_nid, N1S_);
  run_htgt(stream, N1S_, N1D_, E1_, srcbuf, g1_ntype, g1_nid, g1_esrc, g1_edst, g1_etype,
           g1_ew, srcbuf, g1_ntype, convKP + 0 * 4 * PER128, convQP + 0 * 4 * PER128,
           convVP + 0 * 4 * PER128, convOP + 0 * 4 * PER128, convEP + 0 * PERE,
           convMu + 0 * NR_ * HEADS_, dia_w, dia_b, bf_src, bf_dst, bf_ee, kbuf, vbuf,
           qbuf, Ptab, scores, mmax, denb, agg, bf_agg, xg, nullptr, 1);

  // ---- phase D: HGT layer 1 -> x2 (bf16, time-major) ----
  k_tok_transpose_int<<<cdiv(T_, 256), 256, 0, stream>>>(tok_id, seq);
  k_tok_transpose_int<<<cdiv(T_, 256), 256, 0, stream>>>(tok_nt, seq_ntype);
  k_gather_rows<<<cdiv(N2_ * 128, 256), 256, 0, stream>>>(srcbuf, xg, g2_src_sel, N2_);
  k_gather_int<<<cdiv(N2_, 256), 256, 0, stream>>>(gnid2, g1_nid, g2_src_sel, N2_);
  k_gather_int<<<cdiv(N2_, 256), 256, 0, stream>>>(srct2, g1_ntype, g2_src_sel, N2_);
  k_dsth2<<<cdiv(T_ * 128, 256), 256, 0, stream>>>(dsth2, xg, his, tok_sel);
  run_htgt(stream, N2_, T_, E2_, srcbuf, srct2, gnid2, g2_esrc, g2_edst, g2_etype, g2_ew,
           dsth2, tok_nt, convKP + 1 * 4 * PER128, convQP + 1 * 4 * PER128,
           convVP + 1 * 4 * PER128, convOP + 1 * 4 * PER128, convEP + 1 * PERE,
           convMu + 1 * NR_ * HEADS_, dia_w, dia_b, bf_src, bf_dst, bf_ee, kbuf, vbuf,
           qbuf, Ptab, scores, mmax, denb, agg, bf_agg, nullptr, bf_x2, 1);

  // ---- phase E: LSTM2 -> ent (time-major) + h2 output (batch-major) ----
  k_gemm<<<dim3(cdiv(T_, 16), 4), 256, 0, stream>>>(bf_x2, l2WiP, lstm2_b, gates_x,
                                                    nullptr, T_, 512, 128, 128, 512, 0);
  k_fill<<<cdiv(B_ * H_ / 2, 256), 256, 0, stream>>>((float*)hbf, 0.f, B_ * H_ / 2);
  k_fill<<<cdiv(B_ * H_, 256), 256, 0, stream>>>(cbuf, 0.f, B_ * H_);
  for (int l = 0; l < L_; ++l) {
    k_gemm<<<dim3(cdiv(B_, 16), 4), 256, 0, stream>>>(hbf, l2WhP, nullptr, gates_h,
                                                      nullptr, B_, 512, 128, 128, 512, 0);
    k_lstm_step<<<cdiv(B_ * H_, 256), 256, 0, stream>>>(gates_x, gates_h, cbuf, hbf, ent,
                                                        out_h2, l, 1, B_);
  }
  k_f2bf<<<cdiv(T_ * 128, 256), 256, 0, stream>>>(bf_ent, ent, T_ * 128);

  // ---- phase F: token attention pooling -> node_embed / cnt ----
  k_gemm_typed<<<cdiv(T_, 16), 256, 0, stream>>>(bf_ent, ipWP, tok_nt, nullptr, inner,
                                                 nullptr, T_, 2);
  k_row_dot_ia<<<cdiv(T_, 8), 256, 0, stream>>>(abuf, inner, iaW, tok_nt, T_);
  k_fill<<<cdiv(CHILD_, 256), 256, 0, stream>>>(mchild, -__builtin_huge_valf(), CHILD_);
  k_fill<<<cdiv(CHILD_, 256), 256, 0, stream>>>(dchild, 0.f, CHILD_);
  k_fill<<<cdiv(CHILD_, 256), 256, 0, stream>>>(cnt, 0.f, CHILD_);
  k_fill<<<cdiv(CHILD_ * 128, 256), 256, 0, stream>>>(node_embed, 0.f, CHILD_ * 128);
  k_seg_max<<<cdiv(T_, 256), 256, 0, stream>>>(mchild, abuf, tok_id, T_);
  k_seg_exp<<<cdiv(T_, 256), 256, 0, stream>>>(abuf, mchild, dchild, tok_id, T_);
  k_token_accum<<<cdiv(T_ * 128, 256), 256, 0, stream>>>(node_embed, cnt, abuf, dchild,
                                                         ent, tok_id);

  // ---- phase G: x0 + HGT layer 2 -> x3 ----
  k_x0<<<cdiv(N1S_ * 128, 256), 256, 0, stream>>>(srcbuf, emb, node_embed, cnt, hist_emb,
                                                  skip, g1_nid, g1_ntype, N1S_);
  run_htgt(stream, N1S_, N1D_, E1_, srcbuf, g1_ntype, g1_nid, g1_esrc, g1_edst, g1_etype,
           g1_ew, srcbuf, g1_ntype, convKP + 2 * 4 * PER128, convQP + 2 * 4 * PER128,
           convVP + 2 * 4 * PER128, convOP + 2 * 4 * PER128, convEP + 2 * PERE,
           convMu + 2 * NR_ * HEADS_, dia_w, dia_b, bf_src, bf_dst, bf_ee, kbuf, vbuf,
           qbuf, Ptab, scores, mmax, denb, agg, bf_agg, x3, nullptr, 1);

  // ---- phase H: HGT layer 3 -> child_embed (no relu) ----
  run_htgt(stream, N1D_, N4D_, E4_, x3, g1_ntype, g1_nid, g4_esrc, g4_edst, g4_etype,
           g4_ew, x3, g1_ntype, convKP + 3 * 4 * PER128, convQP + 3 * 4 * PER128,
           convVP + 3 * 4 * PER128, convOP + 3 * 4 * PER128, convEP + 3 * PERE,
           convMu + 3 * NR_ * HEADS_, dia_w, dia_b, bf_src, bf_dst, bf_ee, kbuf, vbuf,
           qbuf, Ptab, scores, mmax, denb, agg, bf_agg, out_child, nullptr, 0);

  // ---- phase I: parent mean pooling ----
  k_fill<<<cdiv(PARENT_ * 128, 256), 256, 0, stream>>>(psum, 0.f, PARENT_ * 128);
  k_fill<<<cdiv(PARENT_, 256), 256, 0, stream>>>(pcnt, 0.f, PARENT_);
  k_parent<<<cdiv(EC2P_ * 128, 256), 256, 0, stream>>>(psum, pcnt, out_child, pe_p, pe_c);
  k_parent_div<<<cdiv(PARENT_ * 128, 256), 256, 0, stream>>>(out_parent, psum, pcnt);
}
